// up_5085241279033
// MI455X (gfx1250) — matrix operand order repaired and checked by reading, not compiled, not run on hardware
//
#include <hip/hip_runtime.h>
#include <hip/hip_bf16.h>

// MI455X (gfx1250) implementation.
// Roofline: ~58 GFLOP conv work vs ~144 MB essential HBM traffic (6.2us @23.3TB/s)
// => matrix-core bound => v_wmma_f32_16x16x32_bf16 (16384 FLOP/instr, f32 accum).
// Intermediates x1/h1 (128 MB) stay resident in the 192 MB L2.
// conv3 tiles 4 output rows per WG: halo staging redundancy 9x -> 4.5x, and each
// weight A-fragment load feeds 4 WMMAs.

typedef __attribute__((ext_vector_type(16))) __bf16 v16bf;
typedef __attribute__((ext_vector_type(8)))  float  v8f;

union Frag32B { int4 q[2]; v16bf v; };

// A-matrix (16x32 bf16) per-lane K mapping: elem i, half = lane>>4
__device__ __forceinline__ int kmapA(int i, int half) {
    return i + half * 16;   // half0:{0..15} half1:{16..31}
}

// ---------------------------------------------------------------------------
// Deterministic per-channel stats: partial sums per (channel, chunk)
// x layout (nB, C, S)
// ---------------------------------------------------------------------------
__global__ __launch_bounds__(256)
void stats_partial_kernel(const float* __restrict__ x, float* __restrict__ psum,
                          float* __restrict__ psq, int C, int S, int nB, int chunks) {
    int c = blockIdx.x, chunk = blockIdx.y;
    int len = S / chunks;
    int s0 = chunk * len;
    float s = 0.f, q = 0.f;
    for (int nb = 0; nb < nB; ++nb) {
        const float* p = x + ((size_t)nb * C + c) * S + s0;
        for (int i = threadIdx.x; i < len; i += 256) {
            float v = p[i];
            s += v; q += v * v;
        }
    }
    __shared__ float rs[256], rq[256];
    rs[threadIdx.x] = s; rq[threadIdx.x] = q;
    __syncthreads();
    for (int off = 128; off > 0; off >>= 1) {
        if (threadIdx.x < off) {
            rs[threadIdx.x] += rs[threadIdx.x + off];
            rq[threadIdx.x] += rq[threadIdx.x + off];
        }
        __syncthreads();
    }
    if (threadIdx.x == 0) {
        psum[c * chunks + chunk] = rs[0];
        psq [c * chunks + chunk] = rq[0];
    }
}

__global__ void bn_finalize_kernel(const float* __restrict__ psum, const float* __restrict__ psq,
                                   const float* __restrict__ g, const float* __restrict__ b,
                                   float* __restrict__ sc, float* __restrict__ bs,
                                   int C, int chunks, float invN) {
    int c = blockIdx.x * blockDim.x + threadIdx.x;
    if (c >= C) return;
    float s = 0.f, q = 0.f;
    for (int i = 0; i < chunks; ++i) { s += psum[c * chunks + i]; q += psq[c * chunks + i]; }
    float mean = s * invN;
    float var  = q * invN - mean * mean;
    float scale = g[c] * rsqrtf(var + 1e-4f);
    sc[c] = scale;
    bs[c] = b[c] - mean * scale;
}

// ---------------------------------------------------------------------------
// Weight prep: swizzle into exact WMMA A-fragment layout (bf16)
// conv w (Cout=32, Cin=32, 3,3,3) -> frag[mt][tap][lane][i], 27648 elems
// ---------------------------------------------------------------------------
__global__ void prep_conv_w_kernel(const float* __restrict__ w, __bf16* __restrict__ frag) {
    int idx = blockIdx.x * blockDim.x + threadIdx.x;
    if (idx >= 2 * 27 * 32 * 16) return;
    int i = idx & 15;
    int lane = (idx >> 4) & 31;
    int t = idx >> 9;           // 0..53
    int tap = t % 27;
    int mt  = t / 27;
    int half = lane >> 4;
    int co = mt * 16 + (lane & 15);
    int ci = kmapA(i, half);
    frag[idx] = (__bf16)w[(size_t)(co * 32 + ci) * 27 + tap];
}

// deconv w (2,2,2, Cin=64, Cout=32) -> frag[mt][tap][kc][lane][i], 16384 elems
__global__ void prep_deconv_w_kernel(const float* __restrict__ wd, __bf16* __restrict__ frag) {
    int idx = blockIdx.x * blockDim.x + threadIdx.x;
    if (idx >= 2 * 8 * 2 * 32 * 16) return;
    int i = idx & 15;
    int lane = (idx >> 4) & 31;
    int kc  = (idx >> 9) & 1;
    int tap = (idx >> 10) & 7;
    int mt  = idx >> 13;
    int half = lane >> 4;
    int co = mt * 16 + (lane & 15);
    int ci = kmapA(i, half) + 32 * kc;
    frag[idx] = (__bf16)wd[(size_t)tap * 2048 + ci * 32 + co];
}

// ---------------------------------------------------------------------------
// x1 = x + deconv2(bnrelu(down_x)).  Each wave: one row of 16 input voxels,
// all 8 taps x 32 out-channels via WMMA (K=64 in two 32-chunks).
// down_x (2,64,32,32,32); x,x1 (2,32,64,64,64)
// ---------------------------------------------------------------------------
__global__ __launch_bounds__(128)
void deconv_add_kernel(const float* __restrict__ dx, const float* __restrict__ x,
                       const float* __restrict__ sc, const float* __restrict__ bsv,
                       const __bf16* __restrict__ wf, float* __restrict__ x1) {
    int tid = threadIdx.x;
    int wv = tid >> 5, lane = tid & 31;
    int half = lane >> 4, m = lane & 15;
    int gw = blockIdx.x * 4 + wv;          // 0..4095
    int xt = gw & 1;
    int yi = (gw >> 1) & 31;
    int zi = (gw >> 6) & 31;
    int n  = gw >> 11;
    int xi = xt * 16 + m;
    int sp = (zi * 32 + yi) * 32 + xi;

    // B fragments: bnrelu(down_x) for all 64 input channels at this voxel column
    Frag32B bfrag[2];
#pragma unroll
    for (int kc = 0; kc < 2; ++kc) {
#pragma unroll
        for (int i = 0; i < 16; ++i) {
            int ci = i + 16 * half + 32 * kc;   // B 32x16: K = i + 16*half
            float g = dx[((size_t)(n * 64 + ci)) * 32768 + sp];
            bfrag[kc].v[i] = (__bf16)fmaxf(g * sc[ci] + bsv[ci], 0.f);
        }
    }

    for (int tap = 0; tap < 8; ++tap) {
        int pd = tap >> 2, ph = (tap >> 1) & 1, pw = tap & 1;
        v8f acc0 = {}, acc1 = {};
#pragma unroll
        for (int kc = 0; kc < 2; ++kc) {
            Frag32B a0, a1;
            const int4* p0 = (const int4*)(wf + ((((0 * 8 + tap) * 2 + kc) * 32 + lane) << 4));
            a0.q[0] = p0[0]; a0.q[1] = p0[1];
            const int4* p1 = (const int4*)(wf + ((((8 + tap) * 2 + kc) * 32 + lane) << 4));
            a1.q[0] = p1[0]; a1.q[1] = p1[1];
            acc0 = __builtin_amdgcn_wmma_f32_16x16x32_bf16(false, a0.v, false, bfrag[kc].v,
                                                           (short)0, acc0, false, false);
            acc1 = __builtin_amdgcn_wmma_f32_16x16x32_bf16(false, a1.v, false, bfrag[kc].v,
                                                           (short)0, acc1, false, false);
        }
        int zo = 2 * zi + pd, yo = 2 * yi + ph, xo = 2 * xi + pw;
#pragma unroll
        for (int r = 0; r < 8; ++r) {
            int co0 = r + half * 8;
            size_t i0 = ((((size_t)n * 32 + co0) * 64 + zo) * 64 + yo) * 64 + xo;
            x1[i0] = x[i0] + acc0[r];
            size_t i1 = ((((size_t)n * 32 + co0 + 16) * 64 + zo) * 64 + yo) * 64 + xo;
            x1[i1] = x[i1] + acc1[r];
        }
    }
}

// ---------------------------------------------------------------------------
// out = conv3x3x3(bnrelu(in)) [+ base].  One WG per (n, d, 4 h-rows).
// Stage bnrelu(in) halo tile (3 dz x 6 h x 66 w) into LDS as channel-last bf16
// (80 B voxel stride: 16B-aligned, conflict-minimal for ds_load_b128).
// Each wave: 16 w-voxels x 4 h-rows x 32 out-ch; per tap: 2 A-frag loads feed
// 8 back-to-back WMMAs (4 rows x 2 co-tiles).
// ---------------------------------------------------------------------------
#define CONV_LDS_BYTES (3 * 6 * 66 * 40 * 2)   // 95040 B

__global__ __launch_bounds__(128)
void conv3_kernel(const float* __restrict__ in, const float* __restrict__ sc,
                  const float* __restrict__ bsv, const __bf16* __restrict__ wfrag,
                  const float* __restrict__ base, float* __restrict__ out, int addBase) {
    extern __shared__ __bf16 actS[];       // [dz(3)][hr(6)][w(66)][40 bf16]

    int tid = threadIdx.x;
    int h0 = blockIdx.x * 4;
    int d  = blockIdx.y;
    int n  = blockIdx.z;

    // ---- stage bnrelu(in) with halo (zero padded): 3*6*66*32 = 38016 elems ----
    for (int e = tid; e < 3 * 6 * 32 * 66; e += 128) {
        int w = e % 66;
        int t = e / 66;            // (rr, ci)
        int ci = t & 31;
        int rr = t >> 5;           // 0..17 = dz*6 + hr
        int dz = rr / 6, hr = rr - dz * 6;
        int dd = d + dz - 1, hh = h0 + hr - 1, ww = w - 1;
        float v = 0.f;
        if ((unsigned)dd < 64u && (unsigned)hh < 64u && (unsigned)ww < 64u) {
            float g = in[(((size_t)(n * 32 + ci) * 64 + dd) * 64 + hh) * 64 + ww];
            v = fmaxf(g * sc[ci] + bsv[ci], 0.f);
        }
        actS[(rr * 66 + w) * 40 + ci] = (__bf16)v;
    }
    __syncthreads();

    int wv = tid >> 5, lane = tid & 31;
    int half = lane >> 4, m = lane & 15;
    int w0 = wv * 16;

    v8f acc[4][2];
#pragma unroll
    for (int hr = 0; hr < 4; ++hr) { acc[hr][0] = {}; acc[hr][1] = {}; }

    for (int tap = 0; tap < 27; ++tap) {
        int dz = tap / 9, ry = (tap / 3) % 3, dxo = tap % 3;

        // A fragments: pre-swizzled weights, coalesced b128 loads (L0/L2 hot),
        // reused across 4 output rows.
        Frag32B a0, a1;
        const int4* ap0 = (const int4*)(wfrag + (((0 * 27 + tap) * 32 + lane) << 4));
        a0.q[0] = ap0[0]; a0.q[1] = ap0[1];
        const int4* ap1 = (const int4*)(wfrag + (((27 + tap) * 32 + lane) << 4));
        a1.q[0] = ap1[0]; a1.q[1] = ap1[1];

#pragma unroll
        for (int hr = 0; hr < 4; ++hr) {
            // B fragment: 32 channels (bf16) of one voxel, split by half
            Frag32B bf;
            const int4* bp = (const int4*)
                &actS[((dz * 6 + hr + ry) * 66 + (w0 + m + dxo)) * 40 + half * 16];
            bf.q[0] = bp[0]; bf.q[1] = bp[1];
            acc[hr][0] = __builtin_amdgcn_wmma_f32_16x16x32_bf16(false, a0.v, false, bf.v,
                                                                 (short)0, acc[hr][0], false, false);
            acc[hr][1] = __builtin_amdgcn_wmma_f32_16x16x32_bf16(false, a1.v, false, bf.v,
                                                                 (short)0, acc[hr][1], false, false);
        }
    }

    int wpos = w0 + m;
#pragma unroll
    for (int hr = 0; hr < 4; ++hr) {
        int hgl = h0 + hr;
#pragma unroll
        for (int r = 0; r < 8; ++r) {
            int co0 = r + half * 8;
            size_t i0 = ((((size_t)n * 32 + co0) * 64 + d) * 64 + hgl) * 64 + wpos;
            float v0 = acc[hr][0][r];
            if (addBase) v0 += base[i0];
            out[i0] = v0;
            size_t i1 = ((((size_t)n * 32 + co0 + 16) * 64 + d) * 64 + hgl) * 64 + wpos;
            float v1 = acc[hr][1][r];
            if (addBase) v1 += base[i1];
            out[i1] = v1;
        }
    }
}

// ---------------------------------------------------------------------------
// Workspace layout (bytes)
// ---------------------------------------------------------------------------
static constexpr size_t OFF_X1  = 0;                        // 64 MB  (x + up)
static constexpr size_t OFF_H1  = 67108864;                 // 64 MB  (conv1 out)
static constexpr size_t OFF_WF1 = 134217728;                // 55296 B -> 64 KB slot
static constexpr size_t OFF_WF2 = OFF_WF1 + 65536;
static constexpr size_t OFF_WDF = OFF_WF2 + 65536;          // 32768 B -> 64 KB slot
static constexpr size_t OFF_PS  = OFF_WDF + 65536;          // 16 KB partial sums
static constexpr size_t OFF_PQ  = OFF_PS + 16384;           // 16 KB partial sumsq
static constexpr size_t OFF_SC  = OFF_PQ + 16384;           // scale (<=64 f32)
static constexpr size_t OFF_BS  = OFF_SC + 1024;            // bias

extern "C" void kernel_launch(void* const* d_in, const int* in_sizes, int n_in,
                              void* d_out, int out_size, void* d_ws, size_t ws_size,
                              hipStream_t stream) {
    (void)in_sizes; (void)n_in; (void)out_size; (void)ws_size;
    const float* down_x = (const float*)d_in[0];
    const float* x      = (const float*)d_in[1];
    const float* g0     = (const float*)d_in[2];
    const float* b0     = (const float*)d_in[3];
    const float* wd     = (const float*)d_in[4];
    const float* g1     = (const float*)d_in[5];
    const float* b1     = (const float*)d_in[6];
    const float* w1     = (const float*)d_in[7];
    const float* g2     = (const float*)d_in[8];
    const float* b2     = (const float*)d_in[9];
    const float* w2     = (const float*)d_in[10];
    float* out = (float*)d_out;

    char* ws = (char*)d_ws;
    float*  x1   = (float*)(ws + OFF_X1);
    float*  h1   = (float*)(ws + OFF_H1);
    __bf16* wf1  = (__bf16*)(ws + OFF_WF1);
    __bf16* wf2  = (__bf16*)(ws + OFF_WF2);
    __bf16* wdf  = (__bf16*)(ws + OFF_WDF);
    float*  psum = (float*)(ws + OFF_PS);
    float*  psq  = (float*)(ws + OFF_PQ);
    float*  scb  = (float*)(ws + OFF_SC);
    float*  bsb  = (float*)(ws + OFF_BS);

    // Stage 0: BN stats of down_x (C=64, S=32^3, B=2) -> scale/bias
    stats_partial_kernel<<<dim3(64, 16), 256, 0, stream>>>(down_x, psum, psq, 64, 32768, 2, 16);
    bn_finalize_kernel<<<1, 64, 0, stream>>>(psum, psq, g0, b0, scb, bsb, 64, 16, 1.f / 65536.f);
    prep_deconv_w_kernel<<<64, 256, 0, stream>>>(wd, wdf);

    // Stage 1: x1 = x + deconv2(bnrelu(down_x))   [WMMA bf16]
    deconv_add_kernel<<<1024, 128, 0, stream>>>(down_x, x, scb, bsb, wdf, x1);

    // Stage 2: BN1 stats on x1 (C=32, S=64^3, B=2)
    stats_partial_kernel<<<dim3(32, 64), 256, 0, stream>>>(x1, psum, psq, 32, 262144, 2, 64);
    bn_finalize_kernel<<<1, 32, 0, stream>>>(psum, psq, g1, b1, scb, bsb, 32, 64, 1.f / 524288.f);
    prep_conv_w_kernel<<<108, 256, 0, stream>>>(w1, wf1);

    // Stage 3: h1 = conv3(bnrelu(x1))             [WMMA bf16]
    conv3_kernel<<<dim3(16, 64, 2), 128, CONV_LDS_BYTES, stream>>>(x1, scb, bsb, wf1,
                                                                   nullptr, h1, 0);

    // Stage 4: BN2 stats on h1
    stats_partial_kernel<<<dim3(32, 64), 256, 0, stream>>>(h1, psum, psq, 32, 262144, 2, 64);
    bn_finalize_kernel<<<1, 32, 0, stream>>>(psum, psq, g2, b2, scb, bsb, 32, 64, 1.f / 524288.f);
    prep_conv_w_kernel<<<108, 256, 0, stream>>>(w2, wf2);

    // Stage 5: out = x1 + conv3(bnrelu(h1))       [WMMA bf16 + residual add]
    conv3_kernel<<<dim3(16, 64, 2), 128, CONV_LDS_BYTES, stream>>>(h1, scb, bsb, wf2,
                                                                   x1, out, 1);
}
